// GeometryGuidedAttention_58162447123324
// MI455X (gfx1250) — matrix operand order repaired and checked by reading, not compiled, not run on hardware
//
#include <hip/hip_runtime.h>
#include <math.h>

#define BATCH 2
#define NPTS  16384
#define KNN   16
#define CH    256
#define NHEAD 4
#define HDIM  64
#define MROWS (BATCH * NPTS)   // 32768 points total

typedef __attribute__((ext_vector_type(16))) _Float16 v16h;
typedef __attribute__((ext_vector_type(8)))  _Float16 v8h;
typedef __attribute__((ext_vector_type(8)))  float    v8f;

union V16H { v16h v; v8h h[2]; };

// ---------------------------------------------------------------------------
// WMMA fragment helpers (CDNA5 V_WMMA_F32_16X16X32_F16, wave32)
// A (16x32 f16, row-major [M x ld]): lane m=lane&15 holds row m0+m;
//   lanes<16: elems 0..7 = K k0..k0+7, elems 8..15 = K k0+16..23; lanes>=16: +8.
// B (32x16 f16) from W^T stored row-major WT[n][k]:
//   lane<16: col n0+lane, elems 0..7 = K k0..k0+7, elems 8..15 = K k0+16..23; lanes>=16: +8.
// C/D (16x16 f32): VGPR j = row (j | 8+j by lane half), col = lane&15.
// ---------------------------------------------------------------------------

__device__ inline v16h load_a_frag(const _Float16* A, int ld, int m0, int k0, int lane) {
  int m    = m0 + (lane & 15);
  int koff = (lane & 16) ? 8 : 0;
  const _Float16* p = A + m * ld + k0 + koff;
  V16H u;
  u.h[0] = *(const v8h*)(p);
  u.h[1] = *(const v8h*)(p + 16);
  return u.v;
}

__device__ inline v16h load_b_frag(const _Float16* WT, int ldk, int n0, int k0, int lane) {
  int n    = n0 + (lane & 15);
  int koff = (lane & 16) ? 8 : 0;
  const _Float16* p = WT + n * ldk + k0 + koff;
  V16H u;
  u.h[0] = *(const v8h*)(p);
  u.h[1] = *(const v8h*)(p + 16);
  return u.v;
}

__device__ inline v8f wmma_f16(v16h a, v16h b, v8f c) {
  return __builtin_amdgcn_wmma_f32_16x16x32_f16(
      /*neg_a=*/false, a, /*neg_b=*/false, b,
      /*c_mod=*/(short)0, c, /*reuse_a=*/false, /*reuse_b=*/false);
}

// ---------------------------------------------------------------------------
// CDNA5 async global->LDS copy (ASYNCcnt-tracked fire-and-forget gather).
// lds_off: workgroup-relative LDS byte address (low 32 bits of generic ptr).
// ---------------------------------------------------------------------------

__device__ inline void async_copy_b128(void* lds_ptr, const void* gptr) {
  uint32_t lds_off = (uint32_t)(uintptr_t)lds_ptr;
  uint64_t gaddr   = (uint64_t)(uintptr_t)gptr;
  asm volatile("global_load_async_to_lds_b128 %0, %1, off"
               :: "v"(lds_off), "v"(gaddr) : "memory");
}

__device__ inline void wait_async0() {
  asm volatile("s_wait_asynccnt 0x0" ::: "memory");
}

// ---------------------------------------------------------------------------
// Precision-conversion / weight-packing kernels
// ---------------------------------------------------------------------------

__global__ void cvt_f16_kernel(const float* __restrict__ x, _Float16* __restrict__ y, int n) {
  int i = blockIdx.x * blockDim.x + threadIdx.x;
  if (i < n) y[i] = (_Float16)x[i];
}

// W[rows][cols] (fp32) -> WT[cols][rows] (f16)
__global__ void transpose_cvt_kernel(const float* __restrict__ W, _Float16* __restrict__ WT,
                                     int rows, int cols) {
  int i = blockIdx.x * blockDim.x + threadIdx.x;
  if (i < rows * cols) {
    int n = i / rows;
    int k = i - n * rows;
    WT[n * rows + k] = (_Float16)W[k * cols + n];
  }
}

// ---------------------------------------------------------------------------
// QKV projection: Hq/Hk/Hv[m][c] = sum_k h[m][k] * W[k][c]   (f16 out)
// ---------------------------------------------------------------------------

__global__ __launch_bounds__(256) void qkv_gemm_kernel(
    const _Float16* __restrict__ h16,
    const _Float16* __restrict__ WqT, const _Float16* __restrict__ WkT,
    const _Float16* __restrict__ WvT,
    _Float16* __restrict__ Hq, _Float16* __restrict__ Hk, _Float16* __restrict__ Hv) {
  const int mtiles = MROWS / 16;
  int mt  = blockIdx.x % mtiles;
  int mat = blockIdx.x / mtiles;
  const _Float16* WT = (mat == 0) ? WqT : ((mat == 1) ? WkT : WvT);
  _Float16*       O  = (mat == 0) ? Hq  : ((mat == 1) ? Hk  : Hv);

  int wave = threadIdx.x >> 5, lane = threadIdx.x & 31;
  int m0 = mt * 16;
  int n0 = wave * 32;

  v8f acc0 = {}; v8f acc1 = {};
  for (int k0 = 0; k0 < CH; k0 += 32) {
    v16h a  = load_a_frag(h16, CH, m0, k0, lane);
    v16h b0 = load_b_frag(WT, CH, n0,      k0, lane);
    v16h b1 = load_b_frag(WT, CH, n0 + 16, k0, lane);
    acc0 = wmma_f16(a, b0, acc0);
    acc1 = wmma_f16(a, b1, acc1);
  }
  int col   = lane & 15;
  int rbase = m0 + ((lane & 16) ? 8 : 0);
  for (int j = 0; j < 8; j++) {
    O[(rbase + j) * CH + n0 + col]      = (_Float16)acc0[j];
    O[(rbase + j) * CH + n0 + 16 + col] = (_Float16)acc1[j];
  }
}

// ---------------------------------------------------------------------------
// Fused geometric attention: one workgroup (8 waves) per point.
// Async K/V gather is launched first and overlaps the pe-GEMM.
// ---------------------------------------------------------------------------

__global__ __launch_bounds__(256) void fused_attn_kernel(
    const float* __restrict__ P,        // [B,3,N]
    const int*   __restrict__ idx,      // [B,N,K]
    const float* __restrict__ Wp1,      // [4,C]
    const float* __restrict__ bp1,      // [C]
    const _Float16* __restrict__ Wp2T,  // [C,C]^T f16
    const float* __restrict__ bp2,      // [C]
    const _Float16* __restrict__ Wa1T,  // [64,64]^T f16
    const float* __restrict__ ba1,      // [64]
    const float* __restrict__ Wa2,      // [64]
    const float* __restrict__ ba2,      // [1]
    const _Float16* __restrict__ Hq,
    const _Float16* __restrict__ Hk,
    const _Float16* __restrict__ Hv,
    _Float16* __restrict__ agg) {
  __shared__ int   s_idx[KNN];
  __shared__ float s_geom[KNN][4];
  __shared__ float s_q[CH];
  __shared__ alignas(16) _Float16 s_kn[KNN][CH];    // gathered K rows   8 KB
  __shared__ alignas(16) _Float16 s_vn[KNN][CH];    // gathered V rows   8 KB
  __shared__ alignas(16) _Float16 s_t[KNN][CH];     // relu(geom@Wp1)    8 KB
  __shared__ alignas(16) float    s_pe[KNN][CH];    // pos encoding     16 KB
  __shared__ alignas(16) _Float16 s_rel[KNN][CH];   // Q - K + pe        8 KB
  __shared__ alignas(16) _Float16 s_u[KNN][CH];     // relu(rel@Wa1)     8 KB
  __shared__ alignas(16) float    s_vpe[KNN][CH];   // V + pe           16 KB
  __shared__ float s_logit[KNN][NHEAD];
  __shared__ float s_attn[KNN][NHEAD];

  int pt  = blockIdx.x;          // flat point id in [0, B*N)
  int b   = pt / NPTS;
  int n   = pt - b * NPTS;
  int tid = threadIdx.x;
  int wave = tid >> 5, lane = tid & 31;

  if (tid < KNN) s_idx[tid] = idx[pt * KNN + tid];
  s_q[tid] = (float)Hq[pt * CH + tid];
  __syncthreads();

  // --- launch async K/V gather into LDS (overlaps everything up to stage 4)
  {
    int k = tid >> 4;            // neighbor row 0..15
    int c = (tid & 15) * 16;     // 16-half (32 B) chunk within row
    size_t row = (size_t)(b * NPTS + s_idx[k]) * CH;
    const _Float16* gk = Hk + row + c;
    const _Float16* gv = Hv + row + c;
    async_copy_b128(&s_kn[k][c],     gk);
    async_copy_b128(&s_kn[k][c + 8], gk + 8);
    async_copy_b128(&s_vn[k][c],     gv);
    async_copy_b128(&s_vn[k][c + 8], gv + 8);
  }

  // geom = [rel_pos, |rel_pos|]
  if (tid < KNN) {
    const float* Pb = P + (size_t)b * 3 * NPTS;
    float px = Pb[n], py = Pb[NPTS + n], pz = Pb[2 * NPTS + n];
    int j = s_idx[tid];
    float dx = Pb[j] - px, dy = Pb[NPTS + j] - py, dz = Pb[2 * NPTS + j] - pz;
    s_geom[tid][0] = dx; s_geom[tid][1] = dy; s_geom[tid][2] = dz;
    s_geom[tid][3] = sqrtf(dx * dx + dy * dy + dz * dz);
  }
  __syncthreads();

  // t = relu(geom @ Wp1 + bp1): thread = channel, dot-4 per row
  {
    int c = tid;
    float w0 = Wp1[c], w1 = Wp1[CH + c], w2 = Wp1[2 * CH + c], w3 = Wp1[3 * CH + c];
    float bb = bp1[c];
    for (int k = 0; k < KNN; k++) {
      float acc = fmaf(s_geom[k][0], w0,
                  fmaf(s_geom[k][1], w1,
                  fmaf(s_geom[k][2], w2,
                  fmaf(s_geom[k][3], w3, bb))));
      s_t[k][c] = (_Float16)fmaxf(acc, 0.0f);
    }
  }
  __syncthreads();

  // pe = t @ Wp2 + bp2 : wave covers cols [32w, 32w+32), A loaded once per k-step
  {
    int n0 = wave * 32;
    v8f acc0 = {}; v8f acc1 = {};
    for (int k0 = 0; k0 < CH; k0 += 32) {
      v16h a  = load_a_frag(&s_t[0][0], CH, 0, k0, lane);
      v16h b0 = load_b_frag(Wp2T, CH, n0,      k0, lane);
      v16h b1 = load_b_frag(Wp2T, CH, n0 + 16, k0, lane);
      acc0 = wmma_f16(a, b0, acc0);
      acc1 = wmma_f16(a, b1, acc1);
    }
    int col   = lane & 15;
    int rbase = (lane & 16) ? 8 : 0;
    float bb0 = bp2[n0 + col], bb1 = bp2[n0 + 16 + col];
    for (int j = 0; j < 8; j++) {
      s_pe[rbase + j][n0 + col]      = acc0[j] + bb0;
      s_pe[rbase + j][n0 + 16 + col] = acc1[j] + bb1;
    }
  }

  // gather must have landed before rel/vpe
  wait_async0();
  __syncthreads();

  // rel = Q - K_nbr + pe ; vpe = V_nbr + pe  (all LDS-resident now)
  {
    int c = tid;
    float q = s_q[c];
    for (int k = 0; k < KNN; k++) {
      float kv = (float)s_kn[k][c];
      float vv = (float)s_vn[k][c];
      float pe = s_pe[k][c];
      s_rel[k][c] = (_Float16)(q - kv + pe);
      s_vpe[k][c] = vv + pe;
    }
  }
  __syncthreads();

  // u = relu(rel @ Wa1 + ba1): wave w -> head w>>1, col tiles (w&1)*32 .. +32
  {
    int head  = wave >> 1;
    int cbase = head * HDIM;
    int n0    = (wave & 1) * 32;
    v8f acc0 = {}; v8f acc1 = {};
    for (int k0 = 0; k0 < HDIM; k0 += 32) {
      v16h a  = load_a_frag(&s_rel[0][cbase], CH, 0, k0, lane);
      v16h b0 = load_b_frag(Wa1T, HDIM, n0,      k0, lane);
      v16h b1 = load_b_frag(Wa1T, HDIM, n0 + 16, k0, lane);
      acc0 = wmma_f16(a, b0, acc0);
      acc1 = wmma_f16(a, b1, acc1);
    }
    int col   = lane & 15;
    int rbase = (lane & 16) ? 8 : 0;
    float bb0 = ba1[n0 + col], bb1 = ba1[n0 + 16 + col];
    for (int j = 0; j < 8; j++) {
      s_u[rbase + j][cbase + n0 + col]      = (_Float16)fmaxf(acc0[j] + bb0, 0.0f);
      s_u[rbase + j][cbase + n0 + 16 + col] = (_Float16)fmaxf(acc1[j] + bb1, 0.0f);
    }
  }
  __syncthreads();

  // logits[k][head] = u[k, head*64 : +64] . Wa2 + ba2
  if (tid < KNN * NHEAD) {
    int k = tid >> 2, head = tid & 3;
    const _Float16* up = &s_u[k][head * HDIM];
    float acc = ba2[0];
    for (int d = 0; d < HDIM; d++) acc += (float)up[d] * Wa2[d];
    s_logit[k][head] = acc;
  }
  __syncthreads();

  // softmax over K per head
  if (tid < NHEAD) {
    float mx = -3.4e38f;
    for (int k = 0; k < KNN; k++) mx = fmaxf(mx, s_logit[k][tid]);
    float e[KNN], sum = 0.0f;
    for (int k = 0; k < KNN; k++) { e[k] = __expf(s_logit[k][tid] - mx); sum += e[k]; }
    float inv = 1.0f / sum;
    for (int k = 0; k < KNN; k++) s_attn[k][tid] = e[k] * inv;
  }
  __syncthreads();

  // agg[c] = sum_k attn[k][head(c)] * (V + pe)[k][c]
  {
    int c = tid, head = c >> 6;
    float acc = 0.0f;
    for (int k = 0; k < KNN; k++) acc += s_attn[k][head] * s_vpe[k][c];
    agg[(size_t)pt * CH + c] = (_Float16)acc;
  }
}

// ---------------------------------------------------------------------------
// Output projection: out = agg @ Wo + bo (fp32 out)
// ---------------------------------------------------------------------------

__global__ __launch_bounds__(256) void out_gemm_kernel(
    const _Float16* __restrict__ agg, const _Float16* __restrict__ WoT,
    const float* __restrict__ bo, float* __restrict__ out) {
  int wave = threadIdx.x >> 5, lane = threadIdx.x & 31;
  int m0 = blockIdx.x * 16;
  int n0 = wave * 32;

  v8f acc0 = {}; v8f acc1 = {};
  for (int k0 = 0; k0 < CH; k0 += 32) {
    v16h a  = load_a_frag(agg, CH, m0, k0, lane);
    v16h b0 = load_b_frag(WoT, CH, n0,      k0, lane);
    v16h b1 = load_b_frag(WoT, CH, n0 + 16, k0, lane);
    acc0 = wmma_f16(a, b0, acc0);
    acc1 = wmma_f16(a, b1, acc1);
  }
  int col   = lane & 15;
  int rbase = m0 + ((lane & 16) ? 8 : 0);
  float bb0 = bo[n0 + col], bb1 = bo[n0 + 16 + col];
  for (int j = 0; j < 8; j++) {
    out[(rbase + j) * CH + n0 + col]      = acc0[j] + bb0;
    out[(rbase + j) * CH + n0 + 16 + col] = acc1[j] + bb1;
  }
}

// ---------------------------------------------------------------------------

extern "C" void kernel_launch(void* const* d_in, const int* in_sizes, int n_in,
                              void* d_out, int out_size, void* d_ws, size_t ws_size,
                              hipStream_t stream) {
  const float* h   = (const float*)d_in[0];
  const float* P   = (const float*)d_in[1];
  const int*   idx = (const int*)d_in[2];
  const float* Wq  = (const float*)d_in[3];
  const float* Wk  = (const float*)d_in[4];
  const float* Wv  = (const float*)d_in[5];
  const float* Wp1 = (const float*)d_in[6];
  const float* bp1 = (const float*)d_in[7];
  const float* Wp2 = (const float*)d_in[8];
  const float* bp2 = (const float*)d_in[9];
  const float* Wa1 = (const float*)d_in[10];
  const float* ba1 = (const float*)d_in[11];
  const float* Wa2 = (const float*)d_in[12];
  const float* ba2 = (const float*)d_in[13];
  const float* Wo  = (const float*)d_in[14];
  const float* bo  = (const float*)d_in[15];
  float* out = (float*)d_out;

  // workspace carve-out (all 256B aligned)
  char* ws = (char*)d_ws;
  size_t off = 0;
  auto carve = [&](size_t bytes) -> char* {
    char* p = ws + off;
    off = (off + bytes + 255) & ~(size_t)255;
    return p;
  };
  _Float16* h16  = (_Float16*)carve((size_t)MROWS * CH * 2);
  _Float16* Hq   = (_Float16*)carve((size_t)MROWS * CH * 2);
  _Float16* Hk   = (_Float16*)carve((size_t)MROWS * CH * 2);
  _Float16* Hv   = (_Float16*)carve((size_t)MROWS * CH * 2);
  _Float16* Agg  = (_Float16*)carve((size_t)MROWS * CH * 2);
  _Float16* WqT  = (_Float16*)carve((size_t)CH * CH * 2);
  _Float16* WkT  = (_Float16*)carve((size_t)CH * CH * 2);
  _Float16* WvT  = (_Float16*)carve((size_t)CH * CH * 2);
  _Float16* Wp2T = (_Float16*)carve((size_t)CH * CH * 2);
  _Float16* WoT  = (_Float16*)carve((size_t)CH * CH * 2);
  _Float16* Wa1T = (_Float16*)carve((size_t)HDIM * HDIM * 2);

  // 1) precision conversion + weight packing
  cvt_f16_kernel<<<(MROWS * CH + 255) / 256, 256, 0, stream>>>(h, h16, MROWS * CH);
  transpose_cvt_kernel<<<(CH * CH + 255) / 256, 256, 0, stream>>>(Wq,  WqT,  CH, CH);
  transpose_cvt_kernel<<<(CH * CH + 255) / 256, 256, 0, stream>>>(Wk,  WkT,  CH, CH);
  transpose_cvt_kernel<<<(CH * CH + 255) / 256, 256, 0, stream>>>(Wv,  WvT,  CH, CH);
  transpose_cvt_kernel<<<(CH * CH + 255) / 256, 256, 0, stream>>>(Wp2, Wp2T, CH, CH);
  transpose_cvt_kernel<<<(CH * CH + 255) / 256, 256, 0, stream>>>(Wo,  WoT,  CH, CH);
  transpose_cvt_kernel<<<(HDIM * HDIM + 255) / 256, 256, 0, stream>>>(Wa1, Wa1T, HDIM, HDIM);

  // 2) Q/K/V projections (WMMA)
  qkv_gemm_kernel<<<3 * (MROWS / 16), 256, 0, stream>>>(h16, WqT, WkT, WvT, Hq, Hk, Hv);

  // 3) fused geometric attention, one workgroup per point (WMMA + async gather)
  fused_attn_kernel<<<MROWS, 256, 0, stream>>>(P, idx, Wp1, bp1, Wp2T, bp2,
                                               Wa1T, ba1, Wa2, ba2, Hq, Hk, Hv, Agg);

  // 4) output projection (WMMA)
  out_gemm_kernel<<<MROWS / 16, 256, 0, stream>>>(Agg, WoT, bo, out);

  (void)in_sizes; (void)n_in; (void)out_size; (void)ws_size;
}
